// BondPredictor_51943334478381
// MI455X (gfx1250) — compile-verified, matrix operand order checked by reading, not run on hardware
//
#include <hip/hip_runtime.h>
#include <hip/hip_bf16.h>
#include <math.h>

typedef __attribute__((ext_vector_type(16))) _Float16 v16h;
typedef __attribute__((ext_vector_type(8)))  _Float16 v8h;
typedef __attribute__((ext_vector_type(8)))  float    v8f;
typedef __attribute__((ext_vector_type(4)))  float    v4f;

#define HID 128
#define FRAG_HALFS (100 * 512)            // 100 fragments x 512 f16 = 100 KB
#define FRAG_V8H   (FRAG_HALFS / 8)

// ---------------------------------------------------------------------------
// WMMA fragment layouts (CDNA5 ISA 7.12.2, 16-bit operands, wave32):
//   A (16x32, MxK): lane l -> row M = l%16 ; h = l/16
//     element e (0..15): K = (e<8) ? h*8+e : 16 + h*8 + (e-8)
//   B (32x16, KxN): mirrored: lane l -> col N = l%16, same K(e,h) formula.
//   C/D (16x16 f32): lane l, vgpr v -> M = v + 8*(l/16), N = l%16
//
// Every layer computed as D = W^T (A) x h^T (B):  D lane = edge, vgpr = feature.
// Next layer's B fragment for k-tile t == cat( f16(D_{2t}), f16(D_{2t+1}) ):
// pure in-register hand-off. Weight fragments live in LDS (staged once per
// persistent block) so weight feeds ride DScnt while edge gathers ride LOADcnt.
// ---------------------------------------------------------------------------
__device__ __forceinline__ int k_local(int e, int h) {
    return (e < 8) ? (h * 8 + e) : (16 + h * 8 + (e - 8));
}

// ---------------------------------------------------------------------------
// Kernel 1: pack W^T tiles into WMMA A-fragment layout, f16.
//   f  0..31 : W1 rows   0..127 (source half), f = kt*8 + ft
//   f 32..63 : W1 rows 128..255 (target half)
//   f 64..95 : W2
//   f 96..99 : W3 (single feature tile, features >=4 zero-padded)
// ---------------------------------------------------------------------------
__global__ __launch_bounds__(256) void pack_weights_kernel(
    const float* __restrict__ W1, const float* __restrict__ W2,
    const float* __restrict__ W3, _Float16* __restrict__ Bp)
{
    int gid = blockIdx.x * blockDim.x + threadIdx.x;
    if (gid >= FRAG_HALFS) return;
    int f    = gid >> 9;
    int r    = gid & 511;
    int lane = r >> 4;
    int e    = r & 15;
    int h    = lane >> 4;
    int n    = lane & 15;
    int kl   = k_local(e, h);
    float val;
    if (f < 64) {
        int t = (f & 31) >> 3, nt = f & 7;
        int K = ((f < 32) ? 0 : HID) + t * 32 + kl;
        val = W1[K * HID + nt * 16 + n];
    } else if (f < 96) {
        int f2 = f - 64; int t = f2 >> 3, nt = f2 & 7;
        val = W2[(t * 32 + kl) * HID + nt * 16 + n];
    } else {
        int t = f - 96;
        val = (n < 4) ? W3[(t * 32 + kl) * 4 + n] : 0.0f;
    }
    Bp[gid] = (_Float16)val;
}

// ---------------------------------------------------------------------------
// Kernel 2: node_emb = x@Wa + ba + pos@Wp + bp, stored f16 [N][128]
// ---------------------------------------------------------------------------
__global__ __launch_bounds__(256) void node_embed_kernel(
    const float* __restrict__ x, const float* __restrict__ pos,
    const float* __restrict__ Wa, const float* __restrict__ ba,
    const float* __restrict__ Wp, const float* __restrict__ bp,
    _Float16* __restrict__ emb, int n_nodes)
{
    int gid = blockIdx.x * blockDim.x + threadIdx.x;
    if (gid >= n_nodes * HID) return;
    int node = gid >> 7;
    int j    = gid & 127;
    float s = ba[j] + bp[j];
    const float* xr = x + (size_t)node * 16;
#pragma unroll
    for (int k = 0; k < 16; ++k) s += xr[k] * Wa[k * HID + j];
    const float* pr = pos + (size_t)node * 3;
    s += pr[0] * Wp[j] + pr[1] * Wp[HID + j] + pr[2] * Wp[2 * HID + j];
    emb[gid] = (_Float16)s;
}

// SiLU via v_rcp_f32 (TRANS, co-executes with XDL WMMA), converted to f16
__device__ __forceinline__ v8h silu8_f16(v8f a) {
    v8f r;
#pragma unroll
    for (int i = 0; i < 8; ++i) {
        float xv = a[i];
        r[i] = xv * __builtin_amdgcn_rcpf(1.0f + __expf(-xv));
    }
    return __builtin_convertvector(r, v8h);
}

__device__ __forceinline__ v16h cat8(v8h lo, v8h hi) {
    return __builtin_shufflevector(lo, hi, 0,1,2,3,4,5,6,7,8,9,10,11,12,13,14,15);
}

#define WMMA_F16(A, B, C) \
    __builtin_amdgcn_wmma_f32_16x16x32_f16(false, (A), false, (B), (short)0, (C), false, false)

// ---------------------------------------------------------------------------
// Kernel 3: per-edge MLP. Persistent blocks; one wave = 32 edges
// (2 x 16-edge subtiles), 8 waves/block -> 256 edges/tile-iteration.
// Weight fragments staged once per block into dynamic LDS (100 KB).
// ---------------------------------------------------------------------------
__global__ __launch_bounds__(256) void edge_mlp_kernel(
    const int* __restrict__ ei, const float* __restrict__ pos,
    const _Float16* __restrict__ emb, const _Float16* __restrict__ Bp,
    const float* __restrict__ W1, const float* __restrict__ b1,
    const float* __restrict__ b2, const float* __restrict__ b3,
    float* __restrict__ out, int n_edges, int n_tiles)
{
    extern __shared__ _Float16 lds_w[];          // FRAG_HALFS f16

    // ---- stage the packed weight fragments into LDS once per block --------
    {
        const v8h* g = (const v8h*)Bp;
        v8h*       l = (v8h*)lds_w;
        for (int i = threadIdx.x; i < FRAG_V8H; i += 256) l[i] = g[i];
    }
    __syncthreads();

    const int wave = threadIdx.x >> 5;
    const int lane = threadIdx.x & 31;
    const int h0   = lane >> 4;      // half-wave select
    const int m16  = lane & 15;      // this lane's edge within a subtile
    const v8h* Wl  = (const v8h*)lds_w;          // fragment f, lane l -> Wl[f*64 + l*2]
    const float* w1d = W1 + 256 * HID;           // distance row of W1

    // ---- persistent grid-stride loop over 256-edge block tiles ------------
    for (int tb = blockIdx.x; tb < n_tiles; tb += gridDim.x) {
        const int base = tb * 256 + wave * 32;
        if (base >= n_edges) continue;

        // per-subtile edge gather + distance
        const _Float16* srow[2];
        const _Float16* trow[2];
        float dist[2];
#pragma unroll
        for (int mt = 0; mt < 2; ++mt) {
            int eA = base + mt * 16 + m16;
            if (eA >= n_edges) eA = n_edges - 1;
            int s = ei[eA];
            int d = ei[n_edges + eA];
            srow[mt] = emb + (size_t)s * HID;
            trow[mt] = emb + (size_t)d * HID;
            float dx = pos[(size_t)s * 3 + 0] - pos[(size_t)d * 3 + 0];
            float dy = pos[(size_t)s * 3 + 1] - pos[(size_t)d * 3 + 1];
            float dz = pos[(size_t)s * 3 + 2] - pos[(size_t)d * 3 + 2];
            dist[mt] = sqrtf(dx * dx + dy * dy + dz * dz);
        }

        // ===== Layer 1: D = W1^T x [src|tgt]^T, dist rank-1 in C-init ======
        v8f acc[2][8];                           // [subtile][feature tile]
#pragma unroll
        for (int ft = 0; ft < 8; ++ft) {
            const v4f* bq = (const v4f*)&b1 [ft * 16 + 8 * h0];
            const v4f* wq = (const v4f*)&w1d[ft * 16 + 8 * h0];
            v4f b0 = bq[0], b1v = bq[1], w0 = wq[0], w1v = wq[1];
#pragma unroll
            for (int mt = 0; mt < 2; ++mt)
#pragma unroll
                for (int v = 0; v < 4; ++v) {
                    acc[mt][ft][v]     = b0[v]  + dist[mt] * w0[v];
                    acc[mt][ft][v + 4] = b1v[v] + dist[mt] * w1v[v];
                }
        }
#pragma unroll
        for (int t = 0; t < 4; ++t) {
            // edge-activation B fragments (global/L2, LOADcnt stream)
            v16h bS0 = cat8(*(const v8h*)(srow[0] + t * 32 + h0 * 8),
                            *(const v8h*)(srow[0] + t * 32 + 16 + h0 * 8));
            v16h bS1 = cat8(*(const v8h*)(srow[1] + t * 32 + h0 * 8),
                            *(const v8h*)(srow[1] + t * 32 + 16 + h0 * 8));
            v16h bT0 = cat8(*(const v8h*)(trow[0] + t * 32 + h0 * 8),
                            *(const v8h*)(trow[0] + t * 32 + 16 + h0 * 8));
            v16h bT1 = cat8(*(const v8h*)(trow[1] + t * 32 + h0 * 8),
                            *(const v8h*)(trow[1] + t * 32 + 16 + h0 * 8));
#pragma unroll
            for (int ft = 0; ft < 8; ++ft) {
                // weight fragments from LDS (DScnt stream)
                const v8h* fs = Wl + (size_t)((t * 8 + ft) * 64 + lane * 2);
                const v8h* fr = Wl + (size_t)((32 + t * 8 + ft) * 64 + lane * 2);
                v16h aS = cat8(fs[0], fs[1]);
                v16h aT = cat8(fr[0], fr[1]);
                acc[0][ft] = WMMA_F16(aS, bS0, acc[0][ft]);
                acc[1][ft] = WMMA_F16(aS, bS1, acc[1][ft]);
                acc[0][ft] = WMMA_F16(aT, bT0, acc[0][ft]);
                acc[1][ft] = WMMA_F16(aT, bT1, acc[1][ft]);
            }
        }
        // SiLU + f16, in-register: D feature layout == next-layer B fragment
        v8h h1[2][8];
#pragma unroll
        for (int mt = 0; mt < 2; ++mt)
#pragma unroll
            for (int ft = 0; ft < 8; ++ft) h1[mt][ft] = silu8_f16(acc[mt][ft]);

        // ===== Layer 2: D = W2^T x h1^T ====================================
        v8f acc2[2][8];
#pragma unroll
        for (int ft = 0; ft < 8; ++ft) {
            const v4f* bq = (const v4f*)&b2[ft * 16 + 8 * h0];
            v4f b0 = bq[0], b1v = bq[1];
#pragma unroll
            for (int mt = 0; mt < 2; ++mt)
#pragma unroll
                for (int v = 0; v < 4; ++v) {
                    acc2[mt][ft][v] = b0[v]; acc2[mt][ft][v + 4] = b1v[v];
                }
        }
#pragma unroll
        for (int t = 0; t < 4; ++t) {
            v16h bf0 = cat8(h1[0][2 * t], h1[0][2 * t + 1]);
            v16h bf1 = cat8(h1[1][2 * t], h1[1][2 * t + 1]);
#pragma unroll
            for (int ft = 0; ft < 8; ++ft) {
                const v8h* fw = Wl + (size_t)((64 + t * 8 + ft) * 64 + lane * 2);
                v16h aW = cat8(fw[0], fw[1]);
                acc2[0][ft] = WMMA_F16(aW, bf0, acc2[0][ft]);
                acc2[1][ft] = WMMA_F16(aW, bf1, acc2[1][ft]);
            }
        }
        v8h h2[2][8];
#pragma unroll
        for (int mt = 0; mt < 2; ++mt)
#pragma unroll
            for (int ft = 0; ft < 8; ++ft) h2[mt][ft] = silu8_f16(acc2[mt][ft]);

        // ===== Layer 3: D = W3^T x h2^T (features padded 4->16) ============
        v8f acc3[2];
#pragma unroll
        for (int mt = 0; mt < 2; ++mt)
#pragma unroll
            for (int v = 0; v < 8; ++v)
                acc3[mt][v] = (h0 == 0 && v < 4) ? b3[v] : 0.0f;
#pragma unroll
        for (int t = 0; t < 4; ++t) {
            const v8h* fw = Wl + (size_t)((96 + t) * 64 + lane * 2);
            v16h aW = cat8(fw[0], fw[1]);
            acc3[0] = WMMA_F16(aW, cat8(h2[0][2 * t], h2[0][2 * t + 1]), acc3[0]);
            acc3[1] = WMMA_F16(aW, cat8(h2[1][2 * t], h2[1][2 * t + 1]), acc3[1]);
        }
        // lane = edge, vgprs 0..3 (h0==0) = the 4 logits -> one b128 store/edge
        if (h0 == 0) {
#pragma unroll
            for (int mt = 0; mt < 2; ++mt) {
                int e = base + mt * 16 + m16;
                if (e < n_edges) {
                    v4f o = {acc3[mt][0], acc3[mt][1], acc3[mt][2], acc3[mt][3]};
                    *(v4f*)(out + (size_t)e * 4) = o;
                }
            }
        }
    }
}

// ---------------------------------------------------------------------------
extern "C" void kernel_launch(void* const* d_in, const int* in_sizes, int n_in,
                              void* d_out, int out_size, void* d_ws, size_t ws_size,
                              hipStream_t stream) {
    const float* x   = (const float*)d_in[0];
    const float* pos = (const float*)d_in[1];
    const int*   ei  = (const int*)  d_in[2];
    const float* Wa  = (const float*)d_in[3];
    const float* ba  = (const float*)d_in[4];
    const float* Wp  = (const float*)d_in[5];
    const float* bp  = (const float*)d_in[6];
    const float* W1  = (const float*)d_in[7];
    const float* b1  = (const float*)d_in[8];
    const float* W2  = (const float*)d_in[9];
    const float* b2  = (const float*)d_in[10];
    const float* W3  = (const float*)d_in[11];
    const float* b3  = (const float*)d_in[12];
    float* out = (float*)d_out;

    const int n_nodes = in_sizes[0] / 16;     // ATOM_DIM = 16
    const int n_edges = in_sizes[2] / 2;      // edge_index [2, E]

    // workspace carve-up
    char* ws = (char*)d_ws;
    _Float16* emb = (_Float16*)ws;                                  // N*128 f16
    size_t emb_bytes = (size_t)n_nodes * HID * sizeof(_Float16);
    emb_bytes = (emb_bytes + 255) & ~(size_t)255;
    _Float16* Bp  = (_Float16*)(ws + emb_bytes);                    // FRAG_HALFS f16

    pack_weights_kernel<<<(FRAG_HALFS + 255) / 256, 256, 0, stream>>>(W1, W2, W3, Bp);
    node_embed_kernel<<<((size_t)n_nodes * HID + 255) / 256, 256, 0, stream>>>(
        x, pos, Wa, ba, Wp, bp, emb, n_nodes);

    // persistent blocks: 256 edges per tile-iteration, weights staged in LDS
    int n_tiles  = (n_edges + 255) / 256;
    int n_blocks = n_tiles < 1024 ? n_tiles : 1024;
    size_t lds_bytes = (size_t)FRAG_HALFS * sizeof(_Float16);       // 100 KB
    edge_mlp_kernel<<<n_blocks, 256, lds_bytes, stream>>>(
        ei, pos, emb, Bp, W1, b1, b2, b3, out, n_edges, n_tiles);
}
